// SmallBoardEncoder_40269613367299
// MI455X (gfx1250) — matrix operand order repaired and checked by reading, not compiled, not run on hardware
//
#include <hip/hip_runtime.h>
#include <hip/hip_bf16.h>
#include <math.h>

typedef __attribute__((ext_vector_type(16))) _Float16 v16h;
typedef __attribute__((ext_vector_type(8)))  _Float16 v8h;
typedef __attribute__((ext_vector_type(4)))  _Float16 v4h;
typedef __attribute__((ext_vector_type(8)))  float    v8f;

#define NB     8192
#define NNODE  54
#define NEDGE  144
#define FN     18
#define FE     5
#define HDIM   48
#define OUTD   80
#define NLAYER 3

// mish(x) = x * tanh(softplus(x)) = x * (s-1)/(s+1), s = (1+e^x)^2. Branchless.
__device__ __forceinline__ float mishf(float x) {
  float t = __expf(fminf(x, 20.f));
  float s = 1.f + t;
  s = s * s;
  return x * (s - 1.f) * __builtin_amdgcn_rcpf(s + 1.f);
}

// ---- WMMA fragment helpers (ISA 7.12.2 wave32 layouts) ----
// A (16x32 f16): lane&15 = M; two contiguous 8-half runs per 32-K step at
//   k0 = kt*32+(lane>>4)*8 and k0+16.
// B fragments pre-swizzled in LDS: [kt][nt][lane][16] -> one v16h per lane.
// C/D (f32): col = lane&15; row = r + 8*(lane>>4).
template<int KT, typename FA8>
__device__ __forceinline__ v8f wmma_tile_v(FA8 fa8, const _Float16* wf, int kstride, int lane) {
  v8f c = {};
  const int kh = (lane >> 4) * 8;
#pragma unroll
  for (int kt = 0; kt < KT; ++kt) {
    v8h lo = fa8(kt * 32 + kh);
    v8h hi = fa8(kt * 32 + 16 + kh);
    v16h a;
#pragma unroll
    for (int e = 0; e < 8; ++e) { a[e] = lo[e]; a[8 + e] = hi[e]; }
    v16h b = *(const v16h*)(wf + kt * kstride + lane * 16);
    c = __builtin_amdgcn_wmma_f32_16x16x32_f16(false, a, false, b, (short)0, c, false, false);
  }
  return c;
}

// Scalar-A variant (irregular strides: raw input features).
template<int KT, typename FA>
__device__ __forceinline__ v8f wmma_tile_s(FA fa, const _Float16* wf, int kstride, int lane) {
  v8f c = {};
  const int am = lane & 15;
  const int kh = (lane >> 4) * 8;
#pragma unroll
  for (int kt = 0; kt < KT; ++kt) {
    v16h a;
#pragma unroll
    for (int e = 0; e < 16; ++e) {
      int k = kt * 32 + ((e < 8) ? (kh + e) : (16 + kh + (e - 8)));
      a[e] = fa(am, k);
    }
    v16h b = *(const v16h*)(wf + kt * kstride + lane * 16);
    c = __builtin_amdgcn_wmma_f32_16x16x32_f16(false, a, false, b, (short)0, c, false, false);
  }
  return c;
}

// Stage W[K x N] (f32 row-major, Kreal valid rows) into B-fragment order.
// Linear (k,n) iteration -> coalesced global loads, scattered LDS stores.
// frag index: k -> kt=k/32, e=(k%8)+8*((k/16)&1), half=(k/8)&1; n -> nt=n/16, ln=half*16+n%16
__device__ __forceinline__ void stage_wfrag(_Float16* out, const float* W,
                                            int Kreal, int N, int KT, int NT, int tid) {
  const int total = KT * 32 * N;
  for (int i = tid; i < total; i += 256) {
    int n = i % N, k = i / N;
    float v = (k < Kreal) ? W[k * N + n] : 0.f;
    int kt = k >> 5, e = (k & 7) | ((k >> 1) & 8), half = (k >> 3) & 1;
    int nt = n >> 4, ln = half * 16 + (n & 15);
    out[((kt * NT + nt) * 32 + ln) * 16 + e] = (_Float16)v;
  }
}

// ---------------- Kernel 1: fused per-board GNN ----------------
#define K1_OFF_H    0        // f32 [64][48]   12288
#define K1_OFF_AGG  12288    // f32 [64][48]   12288 (agg, reused as x)
#define K1_OFF_HH   24576    // f16 [64][48]    6144
#define K1_OFF_AGGH 30720    // f16 [64][48]    6144
#define K1_OFF_E    36864    // f16 [144][64]  18432 (cols 48..63 zero)
#define K1_OFF_M1   55296    // f16 [144][64]  18432 (+ staging scratch)
#define K1_OFF_U1   73728    // f16 [64][64]    8192
#define K1_OFF_W    81920    // f16 B-fragments 36864
#define K1_OFF_SRC  118784   // int[144]         576
#define K1_OFF_DST  119360   // int[144]         576
#define K1_OFF_BIAS 119936   // f32[288]        1152
#define K1_SMEM     121088

__global__ __launch_bounds__(256)
void gnn_board_kernel(const float* __restrict__ nodef, const float* __restrict__ edgef,
                      const float* __restrict__ wn, const float* __restrict__ bn,
                      const float* __restrict__ we, const float* __restrict__ be,
                      const float* __restrict__ mw1, const float* __restrict__ mb1,
                      const float* __restrict__ mw2, const float* __restrict__ mb2,
                      const float* __restrict__ uw1, const float* __restrict__ ub1,
                      const float* __restrict__ uw2, const float* __restrict__ ub2,
                      const float* __restrict__ lng, const float* __restrict__ lnb,
                      const int* __restrict__ eidx,
                      float* __restrict__ out_node, float* __restrict__ pooled) {
  extern __shared__ char smem[];
  float*    shH   = (float*)(smem + K1_OFF_H);
  float*    shAgg = (float*)(smem + K1_OFF_AGG);
  _Float16* shHh  = (_Float16*)(smem + K1_OFF_HH);
  _Float16* shAgH = (_Float16*)(smem + K1_OFF_AGGH);
  _Float16* shE   = (_Float16*)(smem + K1_OFF_E);
  _Float16* shM1  = (_Float16*)(smem + K1_OFF_M1);
  _Float16* shU1  = (_Float16*)(smem + K1_OFF_U1);
  _Float16* shW   = (_Float16*)(smem + K1_OFF_W);
  int*      shSrc = (int*)(smem + K1_OFF_SRC);
  int*      shDst = (int*)(smem + K1_OFF_DST);
  float*    shB   = (float*)(smem + K1_OFF_BIAS);
  float*    scrN  = (float*)(smem + K1_OFF_M1);   // temp: raw node feats (972 f)
  float*    scrE  = scrN + 976;                   // temp: raw edge feats (720 f)

  const int tid   = threadIdx.x;
  const int lane  = tid & 31;
  const int wave  = tid >> 5;
  const int board = blockIdx.x;

  // ---- stage inputs (vectorized) + proj weight fragments ----
  {
    const float4* nf4 = (const float4*)(nodef + (size_t)board * NNODE * FN);
    const float4* ef4 = (const float4*)(edgef + (size_t)board * NEDGE * FE);
    float4* sN4 = (float4*)scrN;
    float4* sE4 = (float4*)scrE;
    for (int i = tid; i < (NNODE * FN) / 4; i += 256) sN4[i] = nf4[i];
    for (int i = tid; i < (NEDGE * FE) / 4; i += 256) sE4[i] = ef4[i];
  }
  for (int i = tid; i < NEDGE; i += 256) { shSrc[i] = eidx[i]; shDst[i] = eidx[NEDGE + i]; }
  stage_wfrag(shW,        wn, FN, 48, 1, 3, tid);   // 1536 halfs
  stage_wfrag(shW + 1536, we, FE, 48, 1, 3, tid);   // 1536 halfs
  for (int i = tid; i < 48; i += 256) { shB[i] = bn[i]; shB[48 + i] = be[i]; }
  // zero pad columns of E (whole buffer; real cols rewritten below)
  {
    v8h z = {};
    for (int i = tid; i < (144 * 64) / 8; i += 256) *(v8h*)(shE + i * 8) = z;
  }
  __syncthreads();

  // ---- node projection: h = mish(X @ Wn + bn), M=64(pad) N=48 K=32(pad) ----
  for (int t = wave; t < 12; t += 8) {
    int mt = t / 3, nt = t % 3;
    v8f c = wmma_tile_s<1>(
        [&](int m, int k) -> _Float16 {
          int r = mt * 16 + m;
          return (r < NNODE && k < FN) ? (_Float16)scrN[r * FN + k] : (_Float16)0.f;
        },
        shW + nt * 512, 3 * 512, lane);
    int col  = nt * 16 + (lane & 15);
    float bc = shB[col];
#pragma unroll
    for (int r = 0; r < 8; ++r) {
      int row = mt * 16 + r + 8 * (lane >> 4);
      float v = mishf(c[r] + bc);
      v = (row < NNODE) ? v : 0.f;
      shH[row * 48 + col]  = v;
      shHh[row * 48 + col] = (_Float16)v;
    }
  }
  // ---- edge projection: e = Xe @ We + be (no activation), M=144 ----
  for (int t = wave; t < 27; t += 8) {
    int mt = t / 3, nt = t % 3;
    v8f c = wmma_tile_s<1>(
        [&](int m, int k) -> _Float16 {
          int r = mt * 16 + m;
          return (k < FE) ? (_Float16)scrE[r * FE + k] : (_Float16)0.f;
        },
        shW + 1536 + nt * 512, 3 * 512, lane);
    int col  = nt * 16 + (lane & 15);
    float bc = shB[48 + col];
#pragma unroll
    for (int r = 0; r < 8; ++r) {
      int row = mt * 16 + r + 8 * (lane >> 4);
      shE[row * 64 + col] = (_Float16)(c[r] + bc);
    }
  }
  // zero cols 0..15 of M1 / U1 once (per-layer stores touch only cols < 48)
  {
    v8h z = {};
    for (int i = tid; i < 144 * 2; i += 256) *(v8h*)(shM1 + i * 8 + (i >> 1) * 48) = z;
    // shM1 row r cols 0..15: addresses r*64 and r*64+8, two v8h per row
    for (int i = tid; i < 64 * 2; i += 256)  *(v8h*)(shU1 + i * 8 + (i >> 1) * 48) = z;
  }
  __syncthreads();

  // ---- message-passing layers ----
  _Float16* wf1 = shW;           // msg_w1: KT=5 NT=3 -> 7680
  _Float16* wf2 = shW + 7680;    // msg_w2: KT=2 NT=3 -> 3072
  _Float16* uf1 = shW + 10752;   // upd_w1: KT=3 NT=3 -> 4608
  _Float16* uf2 = shW + 15360;   // upd_w2: KT=2 NT=3 -> 3072

  for (int l = 0; l < NLAYER; ++l) {
    stage_wfrag(wf1, mw1 + l * 144 * 48, 144, 48, 5, 3, tid);
    stage_wfrag(wf2, mw2 + l * 48 * 48,   48, 48, 2, 3, tid);
    stage_wfrag(uf1, uw1 + l * 96 * 48,   96, 48, 3, 3, tid);
    stage_wfrag(uf2, uw2 + l * 48 * 48,   48, 48, 2, 3, tid);
    for (int i = tid; i < 48; i += 256) {
      shB[i]       = mb1[l * 48 + i];
      shB[48 + i]  = mb2[l * 48 + i];
      shB[96 + i]  = ub1[l * 48 + i];
      shB[144 + i] = ub2[l * 48 + i];
      shB[192 + i] = lng[l * 48 + i];
      shB[240 + i] = lnb[l * 48 + i];
    }
    for (int i = tid; i < 64 * 48; i += 256) shAgg[i] = 0.f;
    __syncthreads();

    // msg1: m = mish([h[src],h[dst],e] @ W1 + b1), M=144 N=48 K=160(pad via E cols)
    for (int t = wave; t < 27; t += 8) {
      int mt = t / 3, nt = t % 3;
      int r  = mt * 16 + (lane & 15);
      const _Float16* hs = shHh + shSrc[r] * 48;
      const _Float16* hd = shHh + shDst[r] * 48;
      const _Float16* er = shE + r * 64;
      v8f c = wmma_tile_v<5>(
          [&](int k0) -> v8h {
            const _Float16* p = (k0 < 48) ? (hs + k0)
                              : (k0 < 96) ? (hd + (k0 - 48))
                                          : (er + (k0 - 96));   // e cols 48..63 are zero
            return *(const v8h*)p;
          },
          wf1 + nt * 512, 3 * 512, lane);
      int col  = nt * 16 + (lane & 15);
      float bc = shB[col];
#pragma unroll
      for (int rr = 0; rr < 8; ++rr) {
        int row = mt * 16 + rr + 8 * (lane >> 4);
        shM1[row * 64 + col] = (_Float16)mishf(c[rr] + bc);
      }
    }
    __syncthreads();

    // msg2 + scatter-add by dst (LDS float atomics): K=64 (W2 rows 48..63 are zero)
    for (int t = wave; t < 27; t += 8) {
      int mt = t / 3, nt = t % 3;
      const _Float16* ar = shM1 + (mt * 16 + (lane & 15)) * 64;
      v8f c = wmma_tile_v<2>(
          [&](int k0) -> v8h { return *(const v8h*)(ar + k0); },
          wf2 + nt * 512, 3 * 512, lane);
      int col  = nt * 16 + (lane & 15);
      float bc = shB[48 + col];
#pragma unroll
      for (int rr = 0; rr < 8; ++rr) {
        int row = mt * 16 + rr + 8 * (lane >> 4);
        float v = mishf(c[rr] + bc);
        atomicAdd(&shAgg[shDst[row] * 48 + col], v);
      }
    }
    __syncthreads();

    // f16 mirror of agg for vectorized A loads
    for (int i = tid; i < 64 * 48; i += 256) shAgH[i] = (_Float16)shAgg[i];
    __syncthreads();

    // upd1: u = mish([h, agg] @ U1 + ub1), M=64(pad) K=96
    for (int t = wave; t < 12; t += 8) {
      int mt = t / 3, nt = t % 3;
      int r  = mt * 16 + (lane & 15);
      const _Float16* hr = shHh + r * 48;
      const _Float16* gr = shAgH + r * 48 - 48;
      v8f c = wmma_tile_v<3>(
          [&](int k0) -> v8h {
            const _Float16* p = (k0 < 48) ? (hr + k0) : (gr + k0);
            return *(const v8h*)p;
          },
          uf1 + nt * 512, 3 * 512, lane);
      int col  = nt * 16 + (lane & 15);
      float bc = shB[96 + col];
#pragma unroll
      for (int rr = 0; rr < 8; ++rr) {
        int row = mt * 16 + rr + 8 * (lane >> 4);
        shU1[row * 64 + col] = (_Float16)mishf(c[rr] + bc);
      }
    }
    __syncthreads();

    // upd2 + residual: x = h + u @ U2 + ub2 (into shAgg, reused as x)
    for (int t = wave; t < 12; t += 8) {
      int mt = t / 3, nt = t % 3;
      const _Float16* ur = shU1 + (mt * 16 + (lane & 15)) * 64;
      v8f c = wmma_tile_v<2>(
          [&](int k0) -> v8h { return *(const v8h*)(ur + k0); },
          uf2 + nt * 512, 3 * 512, lane);
      int col  = nt * 16 + (lane & 15);
      float bc = shB[144 + col];
#pragma unroll
      for (int rr = 0; rr < 8; ++rr) {
        int row = mt * 16 + rr + 8 * (lane >> 4);
        shAgg[row * 48 + col] = shH[row * 48 + col] + c[rr] + bc;
      }
    }
    __syncthreads();

    // LayerNorm per node row -> new h (f32 + f16 mirror)
    if (tid < NNODE) {
      float mu = 0.f;
      for (int j = 0; j < 48; ++j) mu += shAgg[tid * 48 + j];
      mu *= (1.f / 48.f);
      float var = 0.f;
      for (int j = 0; j < 48; ++j) { float d = shAgg[tid * 48 + j] - mu; var += d * d; }
      var *= (1.f / 48.f);
      float rs = rsqrtf(var + 1e-5f);
      for (int j = 0; j < 48; ++j) {
        float v = (shAgg[tid * 48 + j] - mu) * rs * shB[192 + j] + shB[240 + j];
        shH[tid * 48 + j]  = v;
        shHh[tid * 48 + j] = (_Float16)v;
      }
    }
    __syncthreads();
  }

  // ---- epilogue: node embeddings (vectorized) + mean/max pool ----
  {
    float4* dst4 = (float4*)(out_node + (size_t)board * NNODE * 48);
    const float4* src4 = (const float4*)shH;
    for (int i = tid; i < (NNODE * 48) / 4; i += 256) dst4[i] = src4[i];
  }
  if (tid < 48) {
    float mu = 0.f, mx = -3.4e38f;
    for (int n = 0; n < NNODE; ++n) {
      float v = shH[n * 48 + tid];
      mu += v;
      mx = fmaxf(mx, v);
    }
    pooled[(size_t)board * 96 + tid]      = mu * (1.f / NNODE);
    pooled[(size_t)board * 96 + 48 + tid] = mx;
  }
}

// ---------------- Kernel 2: batched MLP head over pooled features ----------------
#define K2_OFF_P 0        // f16 [128][96]  24576
#define K2_OFF_W 24576    // f16 B-fragments (KT=3 NT=5) 15360
#define K2_OFF_T 39936    // f16 [128][96]  24576 (t1, cols 80..95 zero)
#define K2_OFF_B 64512    // f32 [160]        640
#define K2_SMEM  65152

__global__ __launch_bounds__(256)
void head_kernel(const float* __restrict__ pooled,
                 const float* __restrict__ w1, const float* __restrict__ b1,
                 const float* __restrict__ w2, const float* __restrict__ b2,
                 float* __restrict__ out) {
  extern __shared__ char smem[];
  _Float16* shP = (_Float16*)(smem + K2_OFF_P);
  _Float16* shW = (_Float16*)(smem + K2_OFF_W);
  _Float16* shT = (_Float16*)(smem + K2_OFF_T);
  float*    shB = (float*)(smem + K2_OFF_B);
  const int tid = threadIdx.x, lane = tid & 31, wave = tid >> 5;
  const int row0 = blockIdx.x * 128;

  {
    const float4* p4 = (const float4*)(pooled + (size_t)row0 * 96);
    for (int i = tid; i < (128 * 96) / 4; i += 256) {
      float4 v = p4[i];
      v4h h; h[0] = (_Float16)v.x; h[1] = (_Float16)v.y;
             h[2] = (_Float16)v.z; h[3] = (_Float16)v.w;
      *(v4h*)(shP + i * 4) = h;
    }
  }
  stage_wfrag(shW, w1, 96, 80, 3, 5, tid);
  {
    v8h z = {};
    for (int i = tid; i < (128 * 96) / 8; i += 256) *(v8h*)(shT + i * 8) = z;
  }
  for (int i = tid; i < 80; i += 256) { shB[i] = b1[i]; shB[80 + i] = b2[i]; }
  __syncthreads();

  // t1 = mish(pooled @ W1 + b1): M=128 N=80 K=96
  for (int t = wave; t < 40; t += 8) {
    int mt = t / 5, nt = t % 5;
    const _Float16* pr = shP + (mt * 16 + (lane & 15)) * 96;
    v8f c = wmma_tile_v<3>(
        [&](int k0) -> v8h { return *(const v8h*)(pr + k0); },
        shW + nt * 512, 5 * 512, lane);
    int cc   = nt * 16 + (lane & 15);
    float bc = shB[cc];
#pragma unroll
    for (int r = 0; r < 8; ++r) {
      int rr = mt * 16 + r + 8 * (lane >> 4);
      shT[rr * 96 + cc] = (_Float16)mishf(c[r] + bc);
    }
  }
  __syncthreads();
  stage_wfrag(shW, w2, 80, 80, 3, 5, tid);   // K padded 80 -> 96
  __syncthreads();

  // out = t1 @ W2 + b2: K=96(pad from 80)
  for (int t = wave; t < 40; t += 8) {
    int mt = t / 5, nt = t % 5;
    const _Float16* tr = shT + (mt * 16 + (lane & 15)) * 96;
    v8f c = wmma_tile_v<3>(
        [&](int k0) -> v8h { return *(const v8h*)(tr + k0); },
        shW + nt * 512, 5 * 512, lane);
    int cc   = nt * 16 + (lane & 15);
    float bc = shB[80 + cc];
#pragma unroll
    for (int r = 0; r < 8; ++r) {
      int rr = row0 + mt * 16 + r + 8 * (lane >> 4);
      out[(size_t)rr * 80 + cc] = c[r] + bc;
    }
  }
}

extern "C" void kernel_launch(void* const* d_in, const int* in_sizes, int n_in,
                              void* d_out, int out_size, void* d_ws, size_t ws_size,
                              hipStream_t stream) {
  const float* nodef = (const float*)d_in[0];
  const float* edgef = (const float*)d_in[1];
  const float* wn    = (const float*)d_in[2];
  const float* bn    = (const float*)d_in[3];
  const float* we    = (const float*)d_in[4];
  const float* be    = (const float*)d_in[5];
  const float* mw1   = (const float*)d_in[6];
  const float* mb1   = (const float*)d_in[7];
  const float* mw2   = (const float*)d_in[8];
  const float* mb2   = (const float*)d_in[9];
  const float* uw1   = (const float*)d_in[10];
  const float* ub1   = (const float*)d_in[11];
  const float* uw2   = (const float*)d_in[12];
  const float* ub2   = (const float*)d_in[13];
  const float* lng   = (const float*)d_in[14];
  const float* lnb   = (const float*)d_in[15];
  const float* wo1   = (const float*)d_in[16];
  const float* bo1   = (const float*)d_in[17];
  const float* wo2   = (const float*)d_in[18];
  const float* bo2   = (const float*)d_in[19];
  const int*   eidx  = (const int*)d_in[20];

  float* out       = (float*)d_out;
  float* out_board = out;                      // [B, 80]
  float* out_node  = out + (size_t)NB * OUTD;  // [B, N, 48]
  float* pooled    = (float*)d_ws;             // [B, 96]

  (void)in_sizes; (void)n_in; (void)out_size; (void)ws_size;

  (void)hipFuncSetAttribute((const void*)gnn_board_kernel,
                            hipFuncAttributeMaxDynamicSharedMemorySize, K1_SMEM);
  (void)hipFuncSetAttribute((const void*)head_kernel,
                            hipFuncAttributeMaxDynamicSharedMemorySize, K2_SMEM);

  gnn_board_kernel<<<NB, 256, K1_SMEM, stream>>>(
      nodef, edgef, wn, bn, we, be, mw1, mb1, mw2, mb2,
      uw1, ub1, uw2, ub2, lng, lnb, eidx, out_node, pooled);

  head_kernel<<<NB / 128, 256, K2_SMEM, stream>>>(pooled, wo1, bo1, wo2, bo2, out_board);
}
